// E71MatrixGatedCUDACell_55456617726100
// MI455X (gfx1250) — matrix operand order repaired and checked by reading, not compiled, not run on hardware
//
#include <hip/hip_runtime.h>
#include <stdint.h>

// Problem constants (match reference)
#define TT 1024
#define BB 32
#define DD 1024
#define NN 64
#define MROWS (TT*BB)        // 32768 rows in the batched projection GEMM

typedef __attribute__((ext_vector_type(16))) __bf16 v16bf;
typedef __attribute__((ext_vector_type(8)))  float  v8f;

// round-to-nearest-even f32 -> bf16, packed pair into one dword
__device__ __forceinline__ uint32_t pack_bf16(float a, float b) {
    uint32_t ua = __float_as_uint(a);
    uint32_t ub = __float_as_uint(b);
    uint32_t ra = (ua + 0x7FFFu + ((ua >> 16) & 1u)) >> 16;
    uint32_t rb = (ub + 0x7FFFu + ((ub >> 16) & 1u)) >> 16;
    return (ra & 0xFFFFu) | (rb << 16);
}

// ---------------------------------------------------------------------------
// Kernel 1: pre-swizzle the 4 weight matrices [64,1024] f32 into bf16
// B-fragment order for v_wmma_f32_16x16x32_bf16, in this code's contiguous k order (matches A).
// B is 32(K) x 16(N): lane n (0-15) holds column n; VGPR r holds
// K = {2r, 2r+1} for lanes 0-15 and K = {16+2r, 17+2r} for lanes 16-31.
// Packed index: (((g*4+nt)*32 + ks)*32 + lane)*8 + r   (one dword = bf16x2)
// ---------------------------------------------------------------------------
__global__ __launch_bounds__(256) void pack_weights(
    const float* __restrict__ Wk, const float* __restrict__ Wv,
    const float* __restrict__ Wq, const float* __restrict__ Wa,
    uint32_t* __restrict__ out) {
    int idx  = blockIdx.x * 256 + threadIdx.x;   // 0 .. 131071
    int r    = idx & 7;
    int lane = (idx >> 3) & 31;
    int ks   = (idx >> 8) & 31;
    int nt   = (idx >> 13) & 3;
    int g    = idx >> 15;
    const float* W = (g == 0) ? Wk : (g == 1) ? Wv : (g == 2) ? Wq : Wa;
    int n   = lane & 15;
    int hi  = lane >> 4;
    int k   = ks * 32 + 2 * r + 16 * hi;
    int row = nt * 16 + n;                       // output-feature index
    out[idx] = pack_bf16(W[row * DD + k], W[row * DD + k + 1]);
}

// ---------------------------------------------------------------------------
// Projection GEMM helpers
// ---------------------------------------------------------------------------
struct BFrag { union { v16bf v; uint32_t u[8]; uint4 q[2]; }; };

__device__ __forceinline__ void load_B16(const uint32_t* base, BFrag* B) {
#pragma unroll
    for (int ntg = 0; ntg < 16; ++ntg) {
        const uint4* bp = (const uint4*)(base + (size_t)ntg * 8192);
        B[ntg].q[0] = bp[0];
        B[ntg].q[1] = bp[1];
    }
}

// lane's A data for one K-step: one contiguous 16-float run
__device__ __forceinline__ void load_A_raw(const float* p, float4* ra) {
    ra[0] = ((const float4*)p)[0];
    ra[1] = ((const float4*)p)[1];
    ra[2] = ((const float4*)(p + 8))[0];
    ra[3] = ((const float4*)(p + 8))[1];
}

__device__ __forceinline__ v16bf cvt_A(const float4* ra) {
    union { v16bf v; uint32_t u[8]; } A;
    A.u[0] = pack_bf16(ra[0].x, ra[0].y);
    A.u[1] = pack_bf16(ra[0].z, ra[0].w);
    A.u[2] = pack_bf16(ra[1].x, ra[1].y);
    A.u[3] = pack_bf16(ra[1].z, ra[1].w);
    A.u[4] = pack_bf16(ra[2].x, ra[2].y);
    A.u[5] = pack_bf16(ra[2].z, ra[2].w);
    A.u[6] = pack_bf16(ra[3].x, ra[3].y);
    A.u[7] = pack_bf16(ra[3].z, ra[3].w);
    return A.v;
}

__device__ __forceinline__ void wmma16(v16bf Av, const BFrag* B, v8f* acc) {
#pragma unroll
    for (int ntg = 0; ntg < 16; ++ntg) {
        acc[ntg] = __builtin_amdgcn_wmma_f32_16x16x32_bf16(
            false, Av, false, B[ntg].v, (short)0, acc[ntg], false, false);
    }
}

// ---------------------------------------------------------------------------
// Kernel 2: fused projection GEMM. One wave per 16-row M tile; 16 C tiles
// (4 projections x 4 N-tiles) accumulated simultaneously so x is read once.
// Software-pipelined one K-step ahead (double-buffered A/B prefetch) with
// sched_barrier(0) pinning loads above the WMMA phase so the scheduler
// cannot re-serialize load->wait->wmma. bf16 f32->pack is deferred to the
// consume phase so no wait targets same-phase loads.
// ---------------------------------------------------------------------------
__global__ __launch_bounds__(128) void proj_gemm(
    const float* __restrict__ x, const uint32_t* __restrict__ wp,
    float* __restrict__ proj) {
    const int wave  = threadIdx.x >> 5;
    const int lane  = threadIdx.x & 31;
    const int mtile = blockIdx.x * 4 + wave;     // 2048 tiles total
    const int m0    = mtile * 16;
    const int mlo   = lane & 15;
    const int hi    = lane >> 4;

    const float*    xa = x + (size_t)(m0 + mlo) * DD + 16 * hi;  // + k0 per step
    const uint32_t* wb = wp + lane * 8;                          // + ks*256, ntg*8192

    v8f acc[16];
#pragma unroll
    for (int i = 0; i < 16; ++i) acc[i] = (v8f){0.f,0.f,0.f,0.f,0.f,0.f,0.f,0.f};

    float4 RA0[4], RA1[4];
    BFrag  B0[16], B1[16];

    load_A_raw(xa, RA0);
    load_B16(wb, B0);

#pragma unroll 1
    for (int ks = 0; ks < 32; ks += 2) {
        // prefetch ks+1
        load_A_raw(xa + (ks + 1) * 32, RA1);
        load_B16(wb + (size_t)(ks + 1) * 256, B1);
        __builtin_amdgcn_sched_barrier(0);
        // consume ks (data has been in flight for a whole phase)
        wmma16(cvt_A(RA0), B0, acc);
        // prefetch ks+2
        if (ks + 2 < 32) {
            load_A_raw(xa + (ks + 2) * 32, RA0);
            load_B16(wb + (size_t)(ks + 2) * 256, B0);
        }
        __builtin_amdgcn_sched_barrier(0);
        // consume ks+1
        wmma16(cvt_A(RA1), B1, acc);
    }

    // ---- store C tiles. C layout: VGPR r, lanes 0-15 -> M=r, N=lane;
    //      lanes 16-31 -> M=8+r, N=lane-16. proj[g] is [M=32768, 64].
#pragma unroll
    for (int ntg = 0; ntg < 16; ++ntg) {
        const int g  = ntg >> 2;
        const int n  = (ntg & 3) * 16 + mlo;
        float* op = proj + (size_t)g * ((size_t)MROWS * NN);
#pragma unroll
        for (int r = 0; r < 8; ++r) {
            int m = m0 + r + 8 * hi;
            op[(size_t)m * NN + n] = acc[ntg][r];
        }
    }
}

// ---------------------------------------------------------------------------
// Kernel 3: sequential gated-matrix scan. One block per batch element.
// Wave w owns rows 8w..8w+7 of S; lane = (row & 7) + 8*chunk, chunk = 16-col
// slice held in 16 VGPRs. Cross-lane reduction via shfl_xor(8/16) only —
// no LDS, no barriers across 1024 steps.
// ---------------------------------------------------------------------------
__global__ __launch_bounds__(256) void scan_kernel(
    const float* __restrict__ proj, const float* __restrict__ S0,
    const float* __restrict__ d_alpha, const float* __restrict__ b_alpha,
    float* __restrict__ out) {
    const int b     = blockIdx.x;
    const int wave  = threadIdx.x >> 5;
    const int lane  = threadIdx.x & 31;
    const int row   = wave * 8 + (lane & 7);
    const int chunk = lane >> 3;
    const int j0    = chunk * 16;

    const size_t PSZ = (size_t)MROWS * NN;
    const float* Kp = proj;
    const float* Vp = proj + PSZ;
    const float* Qp = proj + 2 * PSZ;
    const float* Ap = proj + 3 * PSZ;

    float S[16];
    {
        const float* s0p = S0 + ((size_t)b * NN + row) * NN + j0;
#pragma unroll
        for (int j = 0; j < 16; ++j) S[j] = s0p[j];
    }
    const float dal = d_alpha[row];
    const float bal = b_alpha[row];

    float* outp = out;
    float* sfin = out + (size_t)TT * BB * NN;

    for (int t = 0; t < TT; ++t) {
        const size_t base = ((size_t)t * BB + b) * NN;
        float kk[16], qq[16];
#pragma unroll
        for (int c = 0; c < 4; ++c) {
            float4 fk = *(const float4*)(Kp + base + j0 + c * 4);
            kk[c*4+0] = fk.x; kk[c*4+1] = fk.y; kk[c*4+2] = fk.z; kk[c*4+3] = fk.w;
            float4 fq = *(const float4*)(Qp + base + j0 + c * 4);
            qq[c*4+0] = fq.x; qq[c*4+1] = fq.y; qq[c*4+2] = fq.z; qq[c*4+3] = fq.w;
        }
        const float axv = Ap[base + row];
        const float vv  = Vp[base + row];

        // retrieved = S @ k  (partial per chunk, reduce across 4 chunks)
        float r = 0.f;
#pragma unroll
        for (int j = 0; j < 16; ++j) r = fmaf(S[j], kk[j], r);
        r += __shfl_xor(r, 8);
        r += __shfl_xor(r, 16);

        const float z     = fmaf(dal, r, axv) + bal;
        const float alpha = 1.0f / (1.0f + __expf(-z));
        const float cc    = (1.0f - alpha) * vv;

        // S = alpha*S + (1-alpha)*v*k^T ; h = S_new @ q
        float h = 0.f;
#pragma unroll
        for (int j = 0; j < 16; ++j) {
            S[j] = fmaf(alpha, S[j], cc * kk[j]);
            h    = fmaf(S[j], qq[j], h);
        }
        h += __shfl_xor(h, 8);
        h += __shfl_xor(h, 16);

        if (chunk == 0) {
            const float sg = 1.0f / (1.0f + __expf(-h));
            outp[base + row] = h * h * sg;      // h * silu(h)
        }
    }

    // S_final
    float* sp = sfin + ((size_t)b * NN + row) * NN + j0;
#pragma unroll
    for (int j = 0; j < 16; ++j) sp[j] = S[j];
}

// ---------------------------------------------------------------------------
extern "C" void kernel_launch(void* const* d_in, const int* in_sizes, int n_in,
                              void* d_out, int out_size, void* d_ws, size_t ws_size,
                              hipStream_t stream) {
    const float* x   = (const float*)d_in[0];
    const float* S0  = (const float*)d_in[1];
    const float* Wk  = (const float*)d_in[2];
    const float* Wv  = (const float*)d_in[3];
    const float* Wq  = (const float*)d_in[4];
    const float* Wa  = (const float*)d_in[5];
    const float* dal = (const float*)d_in[6];
    const float* bal = (const float*)d_in[7];

    // workspace: [0, 512KB) packed bf16 weights ; [1MB, 1MB+32MB) projections
    uint32_t* wp   = (uint32_t*)d_ws;
    float*    proj = (float*)((char*)d_ws + (1u << 20));

    pack_weights<<<512, 256, 0, stream>>>(Wk, Wv, Wq, Wa, wp);
    proj_gemm  <<<512, 128, 0, stream>>>(x, wp, proj);
    scan_kernel<<<BB, 256, 0, stream>>>(proj, S0, dal, bal, (float*)d_out);
}
